// Matching_layer_mm_86517821213412
// MI455X (gfx1250) — matrix operand order repaired and checked by reading, not compiled, not run on hardware
//
#include <hip/hip_runtime.h>
#include <hip/hip_bf16.h>

// ---------------------------------------------------------------------------
// Local 9x9 correlation cost volume on gfx1250 via v_wmma_f32_16x16x32_f16.
//
//  Phase 0a: L2-normalize feature1 -> f16, pixel-major [b][pix][64].
//  Phase 0b: L2-normalize feature2 -> f16 into a ZERO-PADDED image
//            [b][72][80][64] (rows -4..67, cols -4..75), so phase 1 needs
//            no bounds checks: OOB candidates load genuine zeros.
//  Phase 1 : per wave32, M=16 query pixels (row segment), K=64 channels
//            (2 x K=32 accumulating WMMA), N=16 candidates x 2 tiles from a
//            single base address (tile 1 = +2048B immediate). Band values
//            go unconditionally to a 16-row LDS stage (rows 9..15 are a
//            dump), then drain as 64B-contiguous global stores.
// ---------------------------------------------------------------------------

typedef _Float16 v8h  __attribute__((ext_vector_type(8)));
typedef _Float16 v16h __attribute__((ext_vector_type(16)));
typedef float    v8f  __attribute__((ext_vector_type(8)));

#define B_    4
#define C_    64
#define H_    64
#define W_    64
#define PATCH 9
#define MAXD  4
#define NPIX  (H_ * W_)        // 4096
#define HP    (H_ + 2 * MAXD)  // 72 padded rows  (-4 .. 67)
#define WP    80               // padded cols -4 .. 75 (covers tile-1 overreach)
#define WAVES_PER_BLOCK 8

// ---------------------------------------------------------------------------
// Phase 0a: feature1 -> f16, pixel-major (no padding needed for queries).
// ---------------------------------------------------------------------------
__global__ void norm_f1_kernel(const float* __restrict__ in,
                               _Float16* __restrict__ out)
{
    int idx = blockIdx.x * blockDim.x + threadIdx.x;   // b*NPIX + pix
    if (idx >= B_ * NPIX) return;
    int b   = idx >> 12;
    int pix = idx & (NPIX - 1);
    const float* f = in + (size_t)b * C_ * NPIX + pix;

    float s = 0.f;
#pragma unroll 8
    for (int c = 0; c < C_; ++c) { float v = f[(size_t)c * NPIX]; s += v * v; }
    float inv = rsqrtf(s + 1e-6f);

    _Float16* o = out + (size_t)idx * C_;
#pragma unroll 8
    for (int c = 0; c < C_; ++c) o[c] = (_Float16)(f[(size_t)c * NPIX] * inv);
}

// ---------------------------------------------------------------------------
// Phase 0b: feature2 -> f16 into padded image; border pixels get zeros.
// ---------------------------------------------------------------------------
__global__ void norm_f2_pad_kernel(const float* __restrict__ in,
                                   _Float16* __restrict__ out)
{
    int idx = blockIdx.x * blockDim.x + threadIdx.x;   // b*HP*WP + rp*WP + cp
    if (idx >= B_ * HP * WP) return;
    int b   = idx / (HP * WP);
    int rem = idx - b * HP * WP;
    int rp  = rem / WP;
    int cp  = rem - rp * WP;
    int i   = rp - MAXD;
    int j   = cp - MAXD;

    _Float16* o = out + (size_t)idx * C_;
    if (i < 0 || i >= H_ || j < 0 || j >= W_) {
#pragma unroll 8
        for (int c = 0; c < C_; ++c) o[c] = (_Float16)0.f;
        return;
    }
    const float* f = in + (size_t)b * C_ * NPIX + (i * W_ + j);
    float s = 0.f;
#pragma unroll 8
    for (int c = 0; c < C_; ++c) { float v = f[(size_t)c * NPIX]; s += v * v; }
    float inv = rsqrtf(s + 1e-6f);
#pragma unroll 8
    for (int c = 0; c < C_; ++c) o[c] = (_Float16)(f[(size_t)c * NPIX] * inv);
}

// Load 16 halfs as two contiguous 16B chunks (lowers to global_load_b128 x2).
__device__ __forceinline__ v16h load_row16(const _Float16* __restrict__ p,
                                           int off0, int off1)
{
    v8h lo = *(const v8h*)(p + off0);
    v8h hi = *(const v8h*)(p + off1);
    v16h r;
#pragma unroll
    for (int t = 0; t < 8; ++t) { r[t] = lo[t]; r[8 + t] = hi[t]; }
    return r;
}

// ---------------------------------------------------------------------------
// Phase 1: one wave32 per (b, row i, 16-col segment j0). 36 WMMAs per wave.
// ---------------------------------------------------------------------------
__global__ void corr_wmma_kernel(const _Float16* __restrict__ f1h,
                                 const _Float16* __restrict__ f2p,
                                 float* __restrict__ out)
{
    // 16 dx-rows x 16 pixels per wave; rows 9..15 are a write-only dump so
    // band extraction needs no predication. Producer == consumer wave, so
    // no barrier (in-wave DScnt ordering suffices).
    __shared__ float stage[WAVES_PER_BLOCK][16 * 16];

    const int lane  = threadIdx.x & 31;
    const int wslot = threadIdx.x >> 5;
    const int wid   = blockIdx.x * (blockDim.x >> 5) + wslot;
    const int seg   = wid & 3;          // 4 segments of 16 cols
    const int i     = (wid >> 2) & 63;  // row
    const int b     = wid >> 8;         // batch
    const int j0    = seg * 16;

    const int nl = lane & 15;
    const int hi = lane >> 4;

    // --- A fragment: 16 query pixels x 64 channels, loaded once per wave.
    // 16-bit A 16x32 layout (ISA 7.12.2): lanes 0-15 carry M=lane with
    // K {0..7,16..23}; lanes 16-31 carry M=lane-16 with K {8..15,24..31}.
    const int kb = hi ? 8 : 0;
    const _Float16* arow =
        f1h + (size_t)(b * NPIX + i * W_ + j0 + nl) * C_;
    v16h a0 = load_row16(arow, kb,      kb + 16);   // channels 0..31
    v16h a1 = load_row16(arow, kb + 32, kb + 48);   // channels 32..63

    // 16-bit B 32x16 layout: lanes 0-15 carry K {0..7,16..23} of column n,
    // lanes 16-31 carry K {8..15,24..31} (same order as A).
    const int koff = hi ? 8 : 0;
    float* outb = out + (size_t)b * (PATCH * PATCH) * NPIX
                      + (size_t)i * W_ + j0;

    for (int q = 0; q < PATCH; ++q) {
        // Padded-image base: row rp = i+q, col cp = j0 + nl (pad cancels the
        // -MAXD). Tile 1 sits +16 pixels = +1024 halfs (immediate offset).
        const _Float16* brow =
            f2p + ((size_t)((b * HP + i + q) * WP + j0 + nl)) * C_ + koff;

        // Speculative prefetch of the next q-row (global_prefetch_b8);
        // clamped so the address stays inside the padded image.
        {
            int rn = i + q + 1; rn = rn > HP - 1 ? HP - 1 : rn;
            __builtin_prefetch(
                f2p + ((size_t)((b * HP + rn) * WP + j0 + nl)) * C_, 0, 1);
        }

        v16h b00 = load_row16(brow, 0,         16);         // t0 ch 0..31
        v16h b01 = load_row16(brow, 32,        48);         // t0 ch 32..63
        v16h b10 = load_row16(brow, 1024,      1024 + 16);  // t1 ch 0..31
        v16h b11 = load_row16(brow, 1024 + 32, 1024 + 48);  // t1 ch 32..63

        v8f acc0 = {}, acc1 = {};
        acc0 = __builtin_amdgcn_wmma_f32_16x16x32_f16(
                   false, a0, false, b00, (short)0, acc0, false, false);
        acc0 = __builtin_amdgcn_wmma_f32_16x16x32_f16(
                   false, a1, false, b01, (short)0, acc0, false, false);
        acc1 = __builtin_amdgcn_wmma_f32_16x16x32_f16(
                   false, a0, false, b10, (short)0, acc1, false, false);
        acc1 = __builtin_amdgcn_wmma_f32_16x16x32_f16(
                   false, a1, false, b11, (short)0, acc1, false, false);

        // Band extraction: element (m, n) lives at lane n (+16 if m>=8),
        // vgpr m&7. d = (n-m) mod 16 selects tile 0 (n>=m) or tile 1 (n<m);
        // valid band is d<9, the rest lands in the LDS dump rows.
#pragma unroll
        for (int r = 0; r < 8; ++r) {
            const int m = r + 8 * hi;
            const int d = (nl - m) & 15;
            const float v = (nl >= m) ? acc0[r] : acc1[r];
            stage[wslot][d * 16 + m] = fmaxf(v, 0.f);
        }

        // Drain 144 band values: 4 full 32-lane stores + 16-lane tail, each
        // covering 64B-contiguous runs of 16 floats per dx row.
#pragma unroll
        for (int it = 0; it < 4; ++it) {
            const int idx = it * 32 + lane;
            const int d = idx >> 4, m = idx & 15;
            outb[(size_t)(q * PATCH + d) * NPIX + m] = stage[wslot][idx];
        }
        if (lane < 16)
            outb[(size_t)(q * PATCH + 8) * NPIX + lane] =
                stage[wslot][128 + lane];
    }
}

extern "C" void kernel_launch(void* const* d_in, const int* in_sizes, int n_in,
                              void* d_out, int out_size, void* d_ws, size_t ws_size,
                              hipStream_t stream)
{
    const float* f1 = (const float*)d_in[0];
    const float* f2 = (const float*)d_in[1];
    float* out = (float*)d_out;

    // ws: f1h = 4*4096*64 halfs (2 MB), f2p = 4*72*80*64 halfs (~2.95 MB)
    _Float16* f1h = (_Float16*)d_ws;
    _Float16* f2p = f1h + (size_t)B_ * NPIX * C_;

    norm_f1_kernel<<<(B_ * NPIX + 255) / 256, 256, 0, stream>>>(f1, f1h);
    norm_f2_pad_kernel<<<(B_ * HP * WP + 255) / 256, 256, 0, stream>>>(f2, f2p);

    // Phase 1: 1024 waves = 128 blocks x 8 waves (256 threads, wave32).
    corr_wmma_kernel<<<128, 256, 0, stream>>>(f1h, f2p, out);
}
